// Transformer_11149735100601
// MI455X (gfx1250) — matrix operand order repaired and checked by reading, not compiled, not run on hardware
//
#include <hip/hip_runtime.h>
#include <hip/hip_bf16.h>
#include <math.h>

// ---------------------------------------------------------------------------
// MI455X (gfx1250) transformer forward.
// HBM-bound problem (~865MB fp32 weights touched once per pass, ~1.4 TFLOP of
// matmul => compute is ~100x below the bf16 WMMA roof). Strategy: stream fp32
// weights once, convert to bf16 with the native packed converter
// (v_cvt_pk_bf16_f32 via __bf16 casts), run v_wmma_f32_16x16x32_bf16 with
// fused bias/relu/residual epilogues to minimize activation traffic.
// ---------------------------------------------------------------------------

#define V_   32000
#define E_   1024
#define DEPTH_ 12
#define H_   16
#define HIDDEN_ 4096
#define B_   2048
#define T_   2
#define HD_  64
#define BT_  (B_ * T_)          // 4096 rows of activations

typedef __bf16 bf16_t;
typedef __attribute__((ext_vector_type(16))) bf16_t v16bf;
typedef __attribute__((ext_vector_type(2)))  bf16_t bf16x2;
typedef __attribute__((ext_vector_type(8)))  float  v8f;
typedef __attribute__((ext_vector_type(4)))  unsigned int u32x4;

union Frag { u32x4 q[2]; v16bf v; };

// Two f32 -> packed bf16 pair (RNE). Lowers to v_cvt_pk_bf16_f32 on gfx1250.
__device__ __forceinline__ unsigned int pack2bf(float a, float b) {
  bf16x2 v;
  v.x = (bf16_t)a;
  v.y = (bf16_t)b;
  return __builtin_bit_cast(unsigned int, v);
}

// ---------------------------------------------------------------------------
// Generic streaming GEMM: C[M,N] = epilogue(A[M,K] @ B[K,N])
//   A: fp32 row-major, leading dim lda
//   B: fp32; bhead==0 -> row-major K x N (ldb); bhead==1 -> head-packed
//      (H, K, 64):  elem(k,n) = B[(n/64)*(K*64) + k*64 + (n%64)]
//   epilogue: +bias[n] (if bias), relu (if relu), +res[m*ldc+n] (if res)
// Tile: 128x128 per workgroup, K-step 32, 8 waves each computing 32x64.
// M, N multiples of 128; K multiple of 32 (true for all GEMMs here).
// ---------------------------------------------------------------------------
#define TM 128
#define TN 128
#define TK 32

__global__ __launch_bounds__(256) void gemm_bf16_wmma(
    const float* __restrict__ A, int lda,
    const float* __restrict__ B, int ldb, int bhead,
    const float* __restrict__ bias, const float* __restrict__ res,
    float* __restrict__ C, int ldc,
    int M, int N, int K, int relu)
{
  __shared__ alignas(16) unsigned short As[2][TM * TK];   // [row][k]
  __shared__ alignas(16) unsigned short Bs[2][TN * TK];   // [col][k] (transposed)

  const int tid    = threadIdx.x;
  const int lane   = tid & 31;
  const int wave   = tid >> 5;        // 0..7
  const int waveM  = wave >> 1;       // 0..3 -> 32 rows each
  const int waveN  = wave & 1;        // 0..1 -> 64 cols each
  const int laneLo = lane & 15;
  const int laneHi = lane >> 4;
  const int m0 = blockIdx.y * TM;
  const int n0 = blockIdx.x * TN;
  const long bheadStride = (long)K * 64;   // head stride for packed QKV weights

  v8f acc[2][4] = {};

  const int KT = K / TK;
  float4 rA[4];            // A: 128x32 as 1024 float4 chunks, 4 per thread
  float4 rB0[2], rB1[2];   // B: 2 chunks of (2 k-rows x 4 n-cols) per thread

  // ---- global load of one K-tile into registers -------------------------
  auto loadTiles = [&](int kt) {
    const int kbase = kt * TK;
#pragma unroll
    for (int i = 0; i < 4; ++i) {
      int c = tid + i * 256;
      int row = c >> 3, col4 = c & 7;
      rA[i] = *(const float4*)(A + (size_t)(m0 + row) * lda + kbase + col4 * 4);
    }
#pragma unroll
    for (int i = 0; i < 2; ++i) {            // 512 chunks: 16 k-pairs x 32 n4
      int c = tid + i * 256;
      int kp = c >> 5, nc4 = c & 31;
      int n = n0 + nc4 * 4;
      int k = kbase + kp * 2;
      const float* p0;
      long kstride;
      if (bhead) {
        p0 = B + (size_t)(n >> 6) * bheadStride + (size_t)k * 64 + (n & 63);
        kstride = 64;
      } else {
        p0 = B + (size_t)k * ldb + n;
        kstride = ldb;
      }
      rB0[i] = *(const float4*)p0;
      rB1[i] = *(const float4*)(p0 + kstride);
      if (kt + 2 < KT)                       // emits global_prefetch_b8
        __builtin_prefetch(p0 + 2 * TK * kstride, 0, 1);
    }
  };

  // ---- convert+store K-tile into LDS ------------------------------------
  auto storeTiles = [&](int buf) {
#pragma unroll
    for (int i = 0; i < 4; ++i) {
      int c = tid + i * 256;
      int row = c >> 3, col4 = c & 7;
      unsigned long long pk =
          (unsigned long long)pack2bf(rA[i].x, rA[i].y) |
          ((unsigned long long)pack2bf(rA[i].z, rA[i].w) << 32);
      *(unsigned long long*)(&As[buf][row * TK + col4 * 4]) = pk;
    }
#pragma unroll
    for (int i = 0; i < 2; ++i) {            // transpose B into [col][k]
      int c = tid + i * 256;
      int kp = c >> 5, nc4 = c & 31;
      int nb = nc4 * 4;
      const float* f0 = (const float*)&rB0[i];
      const float* f1 = (const float*)&rB1[i];
#pragma unroll
      for (int j = 0; j < 4; ++j)            // packed k-pair per ds_store_b32
        *(unsigned int*)(&Bs[buf][(nb + j) * TK + kp * 2]) =
            pack2bf(f0[j], f1[j]);
    }
  };

  loadTiles(0);
  storeTiles(0);
  __syncthreads();

  for (int kt = 0; kt < KT; ++kt) {
    const int cur = kt & 1;
    if (kt + 1 < KT) loadTiles(kt + 1);

    // A fragments (16x32 bf16): lanes 0-15 hold K{0..7,16..23}, lanes 16-31
    // hold K{8..15,24..31} -> two 16B LDS reads at +laneHi*16B and +32B.
    Frag af[2];
#pragma unroll
    for (int mi = 0; mi < 2; ++mi) {
      int row = waveM * 32 + mi * 16 + laneLo;
      const unsigned short* p = &As[cur][row * TK];
      af[mi].q[0] = *(const u32x4*)(p + laneHi * 8);
      af[mi].q[1] = *(const u32x4*)(p + 16 + laneHi * 8);
    }
    // B fragments (32x16 bf16): lanes 0-15 hold K{0..7,16..23}, lanes 16-31 K{8..15,24..31}.
    Frag bfm[4];
#pragma unroll
    for (int ni = 0; ni < 4; ++ni) {
      int col = waveN * 64 + ni * 16 + laneLo;
      const unsigned short* p = &Bs[cur][col * TK + laneHi * 8];
      bfm[ni].q[0] = *(const u32x4*)(p);
      bfm[ni].q[1] = *(const u32x4*)(p + 16);
    }
#pragma unroll
    for (int mi = 0; mi < 2; ++mi)
#pragma unroll
      for (int ni = 0; ni < 4; ++ni)
        acc[mi][ni] = __builtin_amdgcn_wmma_f32_16x16x32_bf16(
            false, af[mi].v, false, bfm[ni].v, (short)0, acc[mi][ni],
            false, false);

    if (kt + 1 < KT) storeTiles(cur ^ 1);
    __syncthreads();
  }

  // ---- epilogue: C/D layout = VGPR i -> M = laneHi*8 + i, N = laneLo -----
#pragma unroll
  for (int mi = 0; mi < 2; ++mi) {
#pragma unroll
    for (int ni = 0; ni < 4; ++ni) {
      int row0 = m0 + waveM * 32 + mi * 16 + laneHi * 8;
      int col  = n0 + waveN * 64 + ni * 16 + laneLo;
      float bv = bias ? bias[col] : 0.0f;
#pragma unroll
      for (int i = 0; i < 8; ++i) {
        size_t idx = (size_t)(row0 + i) * ldc + col;
        float v = acc[mi][ni][i] + bv;
        if (relu) v = fmaxf(v, 0.0f);
        if (res) v += res[idx];
        C[idx] = v;
      }
    }
  }
}

// ---------------------------------------------------------------------------
// Embedding: x[(b*T+t)*E + e] = emb[tok[b*T+t]*E + e] + pos[t*E + e]
// ---------------------------------------------------------------------------
__global__ __launch_bounds__(256) void embed_kernel(
    const int* __restrict__ tokens, const float* __restrict__ emb,
    const float* __restrict__ pos, float* __restrict__ x)
{
  int idx = blockIdx.x * 256 + threadIdx.x;       // over BT_*E_
  int r = idx >> 10;                              // E_ = 1024
  int e = idx & 1023;
  int t = r & (T_ - 1);
  int tok = tokens[r];
  x[idx] = emb[(size_t)tok * E_ + e] + pos[t * E_ + e];
}

// ---------------------------------------------------------------------------
// Attention for T=2: one wave32 per (b,h). Each lane owns dims {l, l+32} of
// HD=64. 2x2 score matrix via __shfl_xor wave reductions, softmax, o = s*v.
// q/k/v/o layout: row (b*T+t), col (h*64 + o), ld = E.
// ---------------------------------------------------------------------------
__global__ __launch_bounds__(256) void attn_kernel(
    const float* __restrict__ q, const float* __restrict__ k,
    const float* __restrict__ v, float* __restrict__ o)
{
  int gwave = (blockIdx.x * 256 + threadIdx.x) >> 5;  // 0 .. B_*H_-1
  int lane  = threadIdx.x & 31;
  int b = gwave >> 4;                                 // H_ = 16
  int h = gwave & 15;
  size_t r0 = (size_t)(b * T_ + 0) * E_ + h * HD_;
  size_t r1 = (size_t)(b * T_ + 1) * E_ + h * HD_;

  float q0a = q[r0 + lane], q0b = q[r0 + lane + 32];
  float q1a = q[r1 + lane], q1b = q[r1 + lane + 32];
  float k0a = k[r0 + lane], k0b = k[r0 + lane + 32];
  float k1a = k[r1 + lane], k1b = k[r1 + lane + 32];
  float v0a = v[r0 + lane], v0b = v[r0 + lane + 32];
  float v1a = v[r1 + lane], v1b = v[r1 + lane + 32];

  float d00 = q0a * k0a + q0b * k0b;
  float d01 = q0a * k1a + q0b * k1b;
  float d10 = q1a * k0a + q1b * k0b;
  float d11 = q1a * k1a + q1b * k1b;
#pragma unroll
  for (int off = 16; off > 0; off >>= 1) {
    d00 += __shfl_xor(d00, off, 32);
    d01 += __shfl_xor(d01, off, 32);
    d10 += __shfl_xor(d10, off, 32);
    d11 += __shfl_xor(d11, off, 32);
  }
  const float scale = 0.125f;   // 1/sqrt(64)
  d00 *= scale; d01 *= scale; d10 *= scale; d11 *= scale;

  float m0 = fmaxf(d00, d01), m1 = fmaxf(d10, d11);
  float e00 = __expf(d00 - m0), e01 = __expf(d01 - m0);
  float e10 = __expf(d10 - m1), e11 = __expf(d11 - m1);
  float p00 = e00 / (e00 + e01), p01 = e01 / (e00 + e01);
  float p10 = e10 / (e10 + e11), p11 = e11 / (e10 + e11);

  o[r0 + lane]      = p00 * v0a + p01 * v1a;
  o[r0 + lane + 32] = p00 * v0b + p01 * v1b;
  o[r1 + lane]      = p10 * v0a + p11 * v1a;
  o[r1 + lane + 32] = p10 * v0b + p11 * v1b;
}

// ---------------------------------------------------------------------------
extern "C" void kernel_launch(void* const* d_in, const int* in_sizes, int n_in,
                              void* d_out, int out_size, void* d_ws, size_t ws_size,
                              hipStream_t stream)
{
  const int*   tokens  = (const int*)  d_in[0];
  const float* emb     = (const float*)d_in[1];
  const float* pos     = (const float*)d_in[2];
  const float* Wq      = (const float*)d_in[3];
  const float* Wk      = (const float*)d_in[4];
  const float* Wv      = (const float*)d_in[5];
  const float* Wo      = (const float*)d_in[6];
  const float* W1      = (const float*)d_in[7];
  const float* b1      = (const float*)d_in[8];
  const float* W2      = (const float*)d_in[9];
  const float* b2      = (const float*)d_in[10];
  const float* unembed = (const float*)d_in[11];
  float* out = (float*)d_out;

  // workspace layout (floats)
  const size_t X = (size_t)BT_ * E_;               // 4096*1024
  float* x     = (float*)d_ws;
  float* qb    = x + X;
  float* kb    = qb + X;
  float* vb    = kb + X;
  float* attno = vb + X;
  float* hbuf  = attno + X;                        // 4096*4096
  const size_t need = (5 * X + (size_t)BT_ * HIDDEN_) * sizeof(float);
  if (ws_size < need) return;

  // --- embedding ----------------------------------------------------------
  embed_kernel<<<(BT_ * E_) / 256, 256, 0, stream>>>(tokens, emb, pos, x);

  const dim3 blk(256);
  const dim3 gridE(E_ / TN, BT_ / TM);             // N=1024 GEMMs
  const dim3 gridH(HIDDEN_ / TN, BT_ / TM);        // N=4096 GEMM
  const size_t wHE  = (size_t)H_ * E_ * HD_;       // per-layer QKV weight stride
  const size_t wEE  = (size_t)E_ * E_;
  const size_t wEH  = (size_t)E_ * HIDDEN_;

  for (int l = 0; l < DEPTH_; ++l) {
    // Q, K, V projections (head-packed weight layout)
    gemm_bf16_wmma<<<gridE, blk, 0, stream>>>(x, E_, Wq + l * wHE, 0, 1,
        nullptr, nullptr, qb, E_, BT_, E_, E_, 0);
    gemm_bf16_wmma<<<gridE, blk, 0, stream>>>(x, E_, Wk + l * wHE, 0, 1,
        nullptr, nullptr, kb, E_, BT_, E_, E_, 0);
    gemm_bf16_wmma<<<gridE, blk, 0, stream>>>(x, E_, Wv + l * wHE, 0, 1,
        nullptr, nullptr, vb, E_, BT_, E_, E_, 0);

    // attention (T=2): B_*H_ waves
    attn_kernel<<<(B_ * H_ * 32) / 256, blk, 0, stream>>>(qb, kb, vb, attno);

    // x = attno @ Wo + x   (in-place residual: each element read+written once)
    gemm_bf16_wmma<<<gridE, blk, 0, stream>>>(attno, E_, Wo + l * wEE, E_, 0,
        nullptr, x, x, E_, BT_, E_, E_, 0);

    // h = relu(x @ W1 + b1)
    gemm_bf16_wmma<<<gridH, blk, 0, stream>>>(x, E_, W1 + l * wEH, HIDDEN_, 0,
        b1 + l * HIDDEN_, nullptr, hbuf, HIDDEN_, BT_, HIDDEN_, E_, 1);

    // x = relu(h @ W2 + b2) + x
    gemm_bf16_wmma<<<gridE, blk, 0, stream>>>(hbuf, HIDDEN_, W2 + l * wEH, E_, 0,
        b2 + l * E_, x, x, E_, BT_, E_, HIDDEN_, 1);
  }

  // logits = x[:, -1, :] @ unembed   (rows b*T + (T-1); lda = T*E)
  const dim3 gridV(V_ / TN, B_ / TM);              // 250 x 16
  gemm_bf16_wmma<<<gridV, blk, 0, stream>>>(x + (T_ - 1) * E_, T_ * E_,
      unembed, V_, 0, nullptr, nullptr, out, V_, B_, V_, E_, 0);
}
